// ODEBlock_73650099192009
// MI455X (gfx1250) — matrix operand order repaired and checked by reading, not compiled, not run on hardware
//
#include <hip/hip_runtime.h>
#include <hip/hip_bf16.h>

typedef __bf16 bf16x16 __attribute__((ext_vector_type(16)));
typedef float  floatx8 __attribute__((ext_vector_type(8)));
typedef unsigned short u16;

#define B_    128
#define C_    64
#define H_    32
#define W_    32
#define HW_   (H_ * W_)            // 1024
#define CHW_  (C_ * HW_)           // 65536
#define NELEM (B_ * CHW_)          // 8388608
#define GROUPS 32
#define GSIZE  ((C_ / GROUPS) * HW_)  // 2048 elements per (b,g)
#define EPS_  1e-5f
#define CIN_  65                   // C+1 (t-channel at ci=0)
#define WSTR  (CIN_ * 9)           // 585, per-out-channel weight stride
#define PACKN (9 * 2 * 4 * 32 * 16)  // 36864 packed bf16 weights per conv

__device__ __forceinline__ u16 f2bf(float f) {
    union { float f; unsigned u; } v; v.f = f;
    unsigned r = v.u + 0x7fffu + ((v.u >> 16) & 1u);
    return (u16)(r >> 16);
}

// ---------------------------------------------------------------------------
// Pack fp32 OIHW conv weights (skipping ci=0) into the per-lane WMMA A-fragment
// layout: pw[tap][cb2][mt][lane][16], lane holds M=lane%16, K-halves by lane/16.
// ---------------------------------------------------------------------------
__global__ __launch_bounds__(256) void k_pack_w(const float* __restrict__ cw,
                                                u16* __restrict__ pw) {
    int idx = blockIdx.x * 256 + threadIdx.x;
    if (idx >= PACKN) return;
    int j    = idx & 15;
    int lane = (idx >> 4) & 31;
    int mt   = (idx >> 9) & 3;
    int cb2  = (idx >> 11) & 1;
    int tap  = idx >> 12;
    int half = lane >> 4;
    int m    = lane & 15;
    int k    = half * 16 + j;
    int o    = mt * 16 + m;
    int ci   = 1 + cb2 * 32 + k;
    pw[idx]  = f2bf(cw[o * WSTR + ci * 9 + tap]);
}

// ---------------------------------------------------------------------------
// GroupNorm stats: one block per (b,g); 2048 contiguous fp32 -> mean, rstd
// ---------------------------------------------------------------------------
__global__ __launch_bounds__(256) void k_gn_stats(const float* __restrict__ x,
                                                  float* __restrict__ stats) {
    int bid = blockIdx.x;               // b*32 + g
    int tid = threadIdx.x;
    const float* p = x + (size_t)bid * GSIZE;
    float s = 0.f, q = 0.f;
    #pragma unroll
    for (int i = 0; i < 8; ++i) {
        float v = p[tid + i * 256];
        s += v; q += v * v;
    }
    #pragma unroll
    for (int off = 16; off > 0; off >>= 1) {
        s += __shfl_xor(s, off, 32);
        q += __shfl_xor(q, off, 32);
    }
    __shared__ float ws[8], wq[8];
    int wv = tid >> 5, lane = tid & 31;
    if (lane == 0) { ws[wv] = s; wq[wv] = q; }
    __syncthreads();
    if (tid == 0) {
        float S = 0.f, Q = 0.f;
        #pragma unroll
        for (int i = 0; i < 8; ++i) { S += ws[i]; Q += wq[i]; }
        float mu  = S * (1.f / GSIZE);
        float var = Q * (1.f / GSIZE) - mu * mu;
        stats[bid * 2 + 0] = mu;
        stats[bid * 2 + 1] = rsqrtf(var + EPS_);
    }
}

// ---------------------------------------------------------------------------
// Apply GN (+ReLU), NCHW fp32 -> NHWC bf16, LDS transpose for coalescing.
// One block per (b,h).
// ---------------------------------------------------------------------------
__global__ __launch_bounds__(256) void k_gn_apply_relu(const float* __restrict__ x,
                                                       const float* __restrict__ stats,
                                                       const float* __restrict__ gw,
                                                       const float* __restrict__ gb,
                                                       u16* __restrict__ act) {
    __shared__ u16 tile[64 * 33];
    int bid = blockIdx.x;               // b*32 + h
    int tid = threadIdx.x;
    int b = bid >> 5, h = bid & 31;
    #pragma unroll
    for (int i = 0; i < 8; ++i) {
        int c = (tid >> 5) + i * 8;
        int w = tid & 31;
        float v   = x[b * CHW_ + c * HW_ + h * 32 + w];
        int   g   = c >> 1;
        float mu  = stats[(b * 32 + g) * 2 + 0];
        float rs  = stats[(b * 32 + g) * 2 + 1];
        float r   = fmaxf(0.f, (v - mu) * rs * gw[c] + gb[c]);
        tile[c * 33 + w] = f2bf(r);
    }
    __syncthreads();
    #pragma unroll
    for (int i = 0; i < 8; ++i) {
        int j = tid + i * 256;          // 0..2047
        int c = j & 63;
        int w = j >> 6;
        act[((size_t)bid * 32 + w) * 64 + c] = tile[c * 33 + w];
    }
}

// ---------------------------------------------------------------------------
// Final GN fused with Euler update: y += dt * gn3(h)   (NCHW fp32 -> NCHW fp32)
// ---------------------------------------------------------------------------
__global__ __launch_bounds__(256) void k_gn3_update(const float* __restrict__ hsrc,
                                                    const float* __restrict__ stats,
                                                    const float* __restrict__ gw,
                                                    const float* __restrict__ gb,
                                                    float* __restrict__ y,
                                                    float dt) {
    int i = blockIdx.x * 256 + threadIdx.x;
    int b = i >> 16;
    int c = (i >> 10) & 63;
    int g = c >> 1;
    float mu = stats[(b * 32 + g) * 2 + 0];
    float rs = stats[(b * 32 + g) * 2 + 1];
    float v  = (hsrc[i] - mu) * rs * gw[c] + gb[c];
    y[i] += dt * v;
}

// ---------------------------------------------------------------------------
// Implicit-GEMM 3x3 conv, bf16 WMMA (f32 accumulate).
//   act : NHWC bf16 [B][32][32][64]
//   pw  : packed A-fragments (k_pack_w layout)
//   cw  : raw fp32 weights (for ci=0 t-channel contribution)
// One block per (b, h-pair): 8 waves, each wave owns a 16(oc) x 32(w) tile of
// one output row, reusing each A fragment for two WMMAs. Out-of-range input
// rows are zero-filled in LDS, so the 36-WMMA main loop is branch-free.
// ---------------------------------------------------------------------------
__global__ __launch_bounds__(256) void k_conv(const u16* __restrict__ act,
                                              const u16* __restrict__ pw,
                                              const float* __restrict__ cw,
                                              const float* __restrict__ bias,
                                              float* __restrict__ outp,
                                              float tval) {
    // 4 rows of 34 padded pixels x 64 ch, bf16 (cols 0 & 33 are zero padding)
    __shared__ __align__(32) u16 sAct[4 * 34 * 64];   // 17408 B
    __shared__ float sW0[64 * 9];

    int bid = blockIdx.x;               // b*16 + hpair
    int tid = threadIdx.x;
    int b  = bid >> 4;
    int h0 = (bid & 15) * 2;

    // warm WGP$/L2 with the packed weight table (global_prefetch_b8)
    __builtin_prefetch((const char*)pw + tid * 288, 0, 1);

    // stage t-channel weights (ci = 0)
    for (int i = tid; i < 576; i += 256)
        sW0[i] = cw[(i / 9) * WSTR + (i % 9)];

    // zero the padded border columns (4 rows x cols {0,33} x 64 ch)
    for (int i = tid; i < 512; i += 256) {
        int r = i >> 7, rem = i & 127;
        int colsel = rem >> 6, cidx = rem & 63;
        sAct[r * 2176 + colsel * 2112 + cidx] = 0;
    }

    // stage activation rows h0-1 .. h0+2; out-of-range rows -> zeros
    #pragma unroll
    for (int r = 0; r < 4; ++r) {
        int hh = h0 - 1 + r;
        uint4* dst = (uint4*)(sAct + r * 2176 + 64);
        if (hh >= 0 && hh < 32) {
            const uint4* src = (const uint4*)(act + ((size_t)(b * 32 + hh) * 32) * 64);
            dst[tid] = src[tid];
        } else {
            uint4 z; z.x = 0; z.y = 0; z.z = 0; z.w = 0;
            dst[tid] = z;
        }
    }
    __syncthreads();

    int wv = tid >> 5, lane = tid & 31;
    int half = lane >> 4, n = lane & 15;
    int mt = wv & 3, nt = wv >> 2;      // nt selects output row h0+nt
    int o0 = mt * 16;
    int h  = h0 + nt;

    const u16* pwbase = pw + (mt << 9) + (lane << 4);

    floatx8 acc0 = {0.f, 0.f, 0.f, 0.f, 0.f, 0.f, 0.f, 0.f};
    floatx8 acc1 = {0.f, 0.f, 0.f, 0.f, 0.f, 0.f, 0.f, 0.f};

    #pragma unroll
    for (int tap = 0; tap < 9; ++tap) {
        int kw    = tap % 3 - 1;
        int rbase = nt + (tap / 3);                    // LDS row = h+kh -(h0-1)
        int off0  = (rbase * 34 + (n + kw + 1)) * 64 + half * 16;   // w = n
        int off1  = off0 + 16 * 64;                                 // w = n+16
        #pragma unroll
        for (int cb2 = 0; cb2 < 2; ++cb2) {
            bf16x16 a  = *(const bf16x16*)(pwbase + (tap * 2 + cb2) * 2048);
            bf16x16 b0 = *(const bf16x16*)(&sAct[off0 + cb2 * 32]);
            bf16x16 b1 = *(const bf16x16*)(&sAct[off1 + cb2 * 32]);
            acc0 = __builtin_amdgcn_wmma_f32_16x16x32_bf16(false, a, false, b0,
                                                           (short)0, acc0, false, false);
            acc1 = __builtin_amdgcn_wmma_f32_16x16x32_bf16(false, a, false, b1,
                                                           (short)0, acc1, false, false);
        }
    }

    // epilogue: bias + analytic t-channel contribution, store NCHW fp32
    union { floatx8 v; float f[8]; } u0, u1;
    u0.v = acc0; u1.v = acc1;
    #pragma unroll
    for (int v = 0; v < 8; ++v) {
        int o = o0 + half * 8 + v;                  // C/D layout: M = v + 8*(lane/16)
        float ts0 = 0.f, ts1 = 0.f;
        #pragma unroll
        for (int tap = 0; tap < 9; ++tap) {
            int kh = tap / 3 - 1, kw = tap % 3 - 1;
            if ((unsigned)(h + kh) < 32u) {
                float wv0 = sW0[o * 9 + tap];
                if ((unsigned)(n + kw) < 32u)      ts0 += wv0;
                if ((unsigned)(n + 16 + kw) < 32u) ts1 += wv0;
            }
        }
        float base = bias[o];
        int   oidx = b * CHW_ + o * HW_ + h * 32;
        outp[oidx + n]      = u0.f[v] + base + tval * ts0;
        outp[oidx + n + 16] = u1.f[v] + base + tval * ts1;
    }
}

// ---------------------------------------------------------------------------
extern "C" void kernel_launch(void* const* d_in, const int* in_sizes, int n_in,
                              void* d_out, int out_size, void* d_ws, size_t ws_size,
                              hipStream_t stream) {
    const float* x       = (const float*)d_in[0];
    const float* gn1_w   = (const float*)d_in[1];
    const float* gn1_b   = (const float*)d_in[2];
    const float* conv1_w = (const float*)d_in[3];
    const float* conv1_b = (const float*)d_in[4];
    const float* gn2_w   = (const float*)d_in[5];
    const float* gn2_b   = (const float*)d_in[6];
    const float* conv2_w = (const float*)d_in[7];
    const float* conv2_b = (const float*)d_in[8];
    const float* gn3_w   = (const float*)d_in[9];
    const float* gn3_b   = (const float*)d_in[10];
    float* y = (float*)d_out;

    char* wsb = (char*)d_ws;
    u16*   pw1   = (u16*)(wsb);                       // 73728 B
    u16*   pw2   = (u16*)(wsb + 73728);               // 73728 B
    float* stats = (float*)(wsb + 147456);            // 32768 B
    u16*   actb  = (u16*)(wsb + 180224);              // 16 MB
    float* hc    = (float*)(wsb + 180224 + (size_t)NELEM * 2);  // 32 MB

    // y = x
    hipMemcpyAsync(y, x, (size_t)NELEM * sizeof(float),
                   hipMemcpyDeviceToDevice, stream);

    // pack conv weights into WMMA A-fragment layout (once per call)
    k_pack_w<<<(PACKN + 255) / 256, 256, 0, stream>>>(conv1_w, pw1);
    k_pack_w<<<(PACKN + 255) / 256, 256, 0, stream>>>(conv2_w, pw2);

    const float dt = 1.0f / 6.0f;
    const int NBG  = B_ * GROUPS;       // 4096
    const int NBH  = B_ * H_;           // 4096
    const int NBH2 = B_ * (H_ / 2);     // 2048 conv workgroups

    for (int step = 0; step < 6; ++step) {
        float t = dt * (float)step;

        // h1 = relu(gn1(y)) -> bf16 NHWC
        k_gn_stats<<<NBG, 256, 0, stream>>>(y, stats);
        k_gn_apply_relu<<<NBH, 256, 0, stream>>>(y, stats, gn1_w, gn1_b, actb);
        // hc = conv1(concat(t, h1))
        k_conv<<<NBH2, 256, 0, stream>>>(actb, pw1, conv1_w, conv1_b, hc, t);
        // h2 = relu(gn2(hc)) -> bf16 NHWC
        k_gn_stats<<<NBG, 256, 0, stream>>>(hc, stats);
        k_gn_apply_relu<<<NBH, 256, 0, stream>>>(hc, stats, gn2_w, gn2_b, actb);
        // hc = conv2(concat(t, h2))
        k_conv<<<NBH2, 256, 0, stream>>>(actb, pw2, conv2_w, conv2_b, hc, t);
        // y += dt * gn3(hc)
        k_gn_stats<<<NBG, 256, 0, stream>>>(hc, stats);
        k_gn3_update<<<NELEM / 256, 256, 0, stream>>>(hc, stats, gn3_w, gn3_b, y, dt);
    }
}
